// WaveletBasis_1606317769441
// MI455X (gfx1250) — matrix operand order repaired and checked by reading, not compiled, not run on hardware
//
#include <hip/hip_runtime.h>
#include <cmath>

// ---------------------------------------------------------------------------
// WaveletBasis fused GEMM for gfx1250 (MI455X).
//   out[b,o] = sum_i x[b,i]*Wb[i,o] + sum_{i,n} basis_n(u[b,i])*C[i,o,n]
// One bf16 WMMA GEMM: M=32768, N=64, K = 1024*8 (wavelet) + 1024 (residual).
// Haar basis computed on the fly per lane (u = sigmoid(2x), branch-free
// exp2+rcp); A/B fragments built directly in registers, no LDS.
// ---------------------------------------------------------------------------

typedef __attribute__((ext_vector_type(16))) __bf16 v16bf;
typedef __attribute__((ext_vector_type(8)))  float  v8f;

union Frag16 { v16bf v; unsigned int u[8]; };

static constexpr int Bsz = 32768;
static constexpr int F   = 1024;
static constexpr int O   = 64;
static constexpr int NB  = 8;

__device__ __forceinline__ unsigned short f2bf(float f) {
  union { __bf16 h; unsigned short s; } r;
  r.h = (__bf16)f;
  return r.s;
}
__device__ __forceinline__ unsigned int pk2bf(float a, float b) {
  return (unsigned int)f2bf(a) | ((unsigned int)f2bf(b) << 16);
}

// 8 Haar basis values of one feature, packed as 4 dwords of bf16 pairs.
// u = (tanh(x)+1)/2 == sigmoid(2x) = 1/(1+exp2(-2*log2e*x)) -> v_exp + v_rcp,
// branch-free (overflow saturates to u=0/1; k3 clamp handles the edges).
// k3 = clamp(floor(8u),0,7) encodes every bucket & sign:
//   level0 sign: bit2, level1 bucket: k3>>2, sign: bit1,
//   level2 bucket: k3>>1, sign: bit0.
__device__ __forceinline__ void basis8(float xv, unsigned int* dst) {
  const float e = __builtin_amdgcn_exp2f(xv * -2.8853900817779268f);
  const float u = __builtin_amdgcn_rcpf(1.0f + e);
  int k3 = (int)(u * 8.0f);
  k3 = k3 < 0 ? 0 : (k3 > 7 ? 7 : k3);
  unsigned short b1 = (k3 & 4) ? 0xBF80u : 0x3F80u;   // -1 / +1
  unsigned short s1 = (k3 & 2) ? 0xBFB5u : 0x3FB5u;   // -/+ sqrt(2) (bf16)
  unsigned short s2 = (k3 & 1) ? 0xC000u : 0x4000u;   // -/+ 2
  const int k1 = k3 >> 2;    // level-1 bucket (0..1)
  const int k2 = k3 >> 1;    // level-2 bucket (0..3)
  unsigned short h2 = (k1 == 0) ? s1 : (unsigned short)0;
  unsigned short h3 = (k1 == 1) ? s1 : (unsigned short)0;
  unsigned short h4 = (k2 == 0) ? s2 : (unsigned short)0;
  unsigned short h5 = (k2 == 1) ? s2 : (unsigned short)0;
  unsigned short h6 = (k2 == 2) ? s2 : (unsigned short)0;
  unsigned short h7 = (k2 == 3) ? s2 : (unsigned short)0;
  dst[0] = 0x3F80u           | ((unsigned int)b1 << 16);  // phi=1, psi0
  dst[1] = (unsigned int)h2  | ((unsigned int)h3 << 16);
  dst[2] = (unsigned int)h4  | ((unsigned int)h5 << 16);
  dst[3] = (unsigned int)h6  | ((unsigned int)h7 << 16);
}

// ---------------------------------------------------------------------------
// Pre-pass: coeffs fp32 [F][O][8] -> bf16 same layout (B-fragment friendly),
// base_weight fp32 [F][O] -> bf16 transposed [O][F], kl = 0.
// ---------------------------------------------------------------------------
__global__ void wavelet_prepass(const float* __restrict__ coeffs,
                                const float* __restrict__ bw,
                                unsigned short* __restrict__ cw,
                                unsigned short* __restrict__ bwT,
                                float* __restrict__ kl) {
  int i = blockIdx.x * blockDim.x + threadIdx.x;
  if (i < F * O * NB) cw[i] = f2bf(coeffs[i]);
  if (i < F * O) {
    int f = i >> 6;      // feature
    int o = i & 63;      // output
    bwT[(size_t)o * F + f] = f2bf(bw[i]);
  }
  if (i == 0) *kl = 0.0f;
}

// ---------------------------------------------------------------------------
// Main fused GEMM. 128 threads = 4 waves; wave -> 32 rows (2 M-tiles of 16)
// x 64 cols (4 N-tiles of 16). Grid 256 -> 32768 rows.
// ---------------------------------------------------------------------------
__global__ __launch_bounds__(128) void wavelet_gemm(
    const float* __restrict__ x,
    const unsigned short* __restrict__ cw,    // bf16 coeffs [F][O][8]
    const unsigned short* __restrict__ bwT,   // bf16 base_weight^T [O][F]
    float* __restrict__ out) {
  const int lane = threadIdx.x & 31;
  const int wave = threadIdx.x >> 5;
  const int half = lane >> 4;     // lane group (0: lanes 0-15, 1: 16-31)
  const int mrow = lane & 15;
  const int row0 = blockIdx.x * 128 + wave * 32;

  v8f acc[2][4] = {};  // f32 accumulators: 2 M-tiles x 4 N-tiles

  // Per-M-tile x row pointers for this lane.
  const float* xr0 = x + (size_t)(row0 + mrow) * F;
  const float* xr1 = x + (size_t)(row0 + 16 + mrow) * F;

  // ---- wavelet part: K = F*8, chunk of 32 = 4 features -------------------
  // Per chunk c: one float4 per M-tile covers features 4c..4c+3; this lane
  // uses elements [half] and [half+2]. Double-buffered across chunks.
  constexpr int NC = F / 4;  // 256
  float4 xc0 = *(const float4*)(xr0);
  float4 xc1 = *(const float4*)(xr1);
#pragma unroll 1
  for (int c = 0; c < NC; ++c) {
    const int cn = (c + 1 < NC) ? (c + 1) : c;  // prefetch next (clamped)
    const float4 xn0 = *(const float4*)(xr0 + 4 * cn);
    const float4 xn1 = *(const float4*)(xr1 + 4 * cn);

    Frag16 a[2];
    {
      const float a0 = half ? xc0.y : xc0.x;   // feature 4c+half
      const float a1 = half ? xc0.w : xc0.z;   // feature 4c+2+half
      basis8(a0, &a[0].u[0]);
      basis8(a1, &a[0].u[4]);
      const float b0 = half ? xc1.y : xc1.x;
      const float b1 = half ? xc1.w : xc1.z;
      basis8(b0, &a[1].u[0]);
      basis8(b1, &a[1].u[4]);
    }

    // B fragments: lane holds coeffs of features g1 (K-elems 0-7), g2 (8-15)
    const int g1 = 4 * c + half;
    const int g2 = g1 + 2;
#pragma unroll
    for (int tn = 0; tn < 4; ++tn) {
      const int o = tn * 16 + mrow;
      Frag16 bfr;
      const uint4 t0 = *(const uint4*)(cw + ((size_t)g1 * O + o) * NB);
      const uint4 t1 = *(const uint4*)(cw + ((size_t)g2 * O + o) * NB);
      bfr.u[0] = t0.x; bfr.u[1] = t0.y; bfr.u[2] = t0.z; bfr.u[3] = t0.w;
      bfr.u[4] = t1.x; bfr.u[5] = t1.y; bfr.u[6] = t1.z; bfr.u[7] = t1.w;
#pragma unroll
      for (int tm = 0; tm < 2; ++tm) {
        acc[tm][tn] = __builtin_amdgcn_wmma_f32_16x16x32_bf16(
            false, a[tm].v, false, bfr.v, (short)0, acc[tm][tn], false, false);
      }
    }
    xc0 = xn0;
    xc1 = xn1;
  }

  // ---- residual part: K = F, chunk of 32 features ------------------------
#pragma unroll 1
  for (int c = 0; c < F / 32; ++c) {
    const int kb = c * 32 + half * 8;  // this lane-group's first K run
    Frag16 a[2];
#pragma unroll
    for (int tm = 0; tm < 2; ++tm) {
      const float* xp = (tm ? xr1 : xr0) + kb;
      const float4 v0 = *(const float4*)(xp + 0);
      const float4 v1 = *(const float4*)(xp + 4);
      const float4 v2 = *(const float4*)(xp + 16);
      const float4 v3 = *(const float4*)(xp + 20);
      a[tm].u[0] = pk2bf(v0.x, v0.y);  a[tm].u[1] = pk2bf(v0.z, v0.w);
      a[tm].u[2] = pk2bf(v1.x, v1.y);  a[tm].u[3] = pk2bf(v1.z, v1.w);
      a[tm].u[4] = pk2bf(v2.x, v2.y);  a[tm].u[5] = pk2bf(v2.z, v2.w);
      a[tm].u[6] = pk2bf(v3.x, v3.y);  a[tm].u[7] = pk2bf(v3.z, v3.w);
    }
#pragma unroll
    for (int tn = 0; tn < 4; ++tn) {
      const int o = tn * 16 + mrow;
      const unsigned short* bp = bwT + (size_t)o * F + c * 32 + half * 8;
      Frag16 bfr;
      const uint4 t0 = *(const uint4*)(bp + 0);
      const uint4 t1 = *(const uint4*)(bp + 16);
      bfr.u[0] = t0.x; bfr.u[1] = t0.y; bfr.u[2] = t0.z; bfr.u[3] = t0.w;
      bfr.u[4] = t1.x; bfr.u[5] = t1.y; bfr.u[6] = t1.z; bfr.u[7] = t1.w;
#pragma unroll
      for (int tm = 0; tm < 2; ++tm) {
        acc[tm][tn] = __builtin_amdgcn_wmma_f32_16x16x32_bf16(
            false, a[tm].v, false, bfr.v, (short)0, acc[tm][tn], false, false);
      }
    }
  }

  // ---- epilogue: C/D layout VGPR r -> (M = r + 8*half, N = mrow) ---------
#pragma unroll
  for (int tm = 0; tm < 2; ++tm) {
#pragma unroll
    for (int tn = 0; tn < 4; ++tn) {
      const int col   = tn * 16 + mrow;
      const int rbase = row0 + tm * 16 + half * 8;
#pragma unroll
      for (int r = 0; r < 8; ++r) {
        out[(size_t)(rbase + r) * O + col] = acc[tm][tn][r];
      }
    }
  }
}

// ---------------------------------------------------------------------------
extern "C" void kernel_launch(void* const* d_in, const int* in_sizes, int n_in,
                              void* d_out, int out_size, void* d_ws, size_t ws_size,
                              hipStream_t stream) {
  const float* x      = (const float*)d_in[0];   // [B, F] fp32
  const float* coeffs = (const float*)d_in[1];   // [F, O, 8] fp32
  const float* bw     = (const float*)d_in[2];   // [F, O] fp32
  float* out = (float*)d_out;                    // [B*O] out ++ [1] kl

  // Workspace layout (needs 1,179,648 bytes):
  //   [0 .. 1MB)       bf16 coeffs  [F][O][8]
  //   [1MB .. +128KB)  bf16 base_weight^T [O][F]
  unsigned short* cw  = (unsigned short*)d_ws;
  unsigned short* bwT = cw + (size_t)F * O * NB;

  const int prepN = F * O * NB;  // 524288 dominates
  wavelet_prepass<<<(prepN + 255) / 256, 256, 0, stream>>>(
      coeffs, bw, cw, bwT, out + (size_t)Bsz * O);

  wavelet_gemm<<<Bsz / 128, 128, 0, stream>>>(x, cw, bwT, out);
}
